// QLinear_89300960018970
// MI455X (gfx1250) — 8-bit matrix operand order repaired and checked by reading, compiled, not run on hardware
//
#include <hip/hip_runtime.h>
#include <cstdint>

typedef __attribute__((ext_vector_type(8))) int v8i;

#define QMINF (-128.0f)
#define QMAXF (127.0f)
#define SMALL_SCALE 6.1e-05f

#define BM 128
#define BN 128
#define BK 64
#define LDST 80
#define TILE_BYTES (128 * LDST)

__global__ __launch_bounds__(256) void quant_rows_kernel(
    const float* __restrict__ x, int8_t* __restrict__ q,
    float* __restrict__ scales, int* __restrict__ zps, int* __restrict__ sums,
    int M, int K)
{
    const int wave = threadIdx.x >> 5;
    const int lane = threadIdx.x & 31;
    const int row  = blockIdx.x * 8 + wave;
    if (row >= M) return;

    const float4* xrow = (const float4*)(x + (size_t)row * K);
    const int K4 = K >> 2;
    const int nch = K4 >> 5;

    float4 vals[8];
    float vmin = 0.0f, vmax = 0.0f, vsum = 0.0f;
#pragma unroll
    for (int j = 0; j < 8; ++j) {
        if (j >= nch) break;
        float4 v = xrow[lane + (j << 5)];
        vals[j] = v;
        vmin = fminf(vmin, fminf(fminf(v.x, v.y), fminf(v.z, v.w)));
        vmax = fmaxf(vmax, fmaxf(fmaxf(v.x, v.y), fmaxf(v.z, v.w)));
        vsum += (v.x + v.y) + (v.z + v.w);
    }
#pragma unroll
    for (int off = 16; off >= 1; off >>= 1) {
        vmin = fminf(vmin, __shfl_xor(vmin, off, 32));
        vmax = fmaxf(vmax, __shfl_xor(vmax, off, 32));
        vsum += __shfl_xor(vsum, off, 32);
    }

    float rmin = vmin, rmax = vmax;
    float scale = (rmax - rmin) / (QMAXF - QMINF);
    float cmpv = __builtin_isinf(1.0f / scale) ? 1.0f : 0.0f;
    if (scale == cmpv) scale = 0.1f;
    bool small = scale < SMALL_SCALE;
    float amp = SMALL_SCALE / scale;
    if (small) { rmax *= amp; rmin *= amp; scale = SMALL_SCALE; }
    float inv = 1.0f / scale;

    float zmin = QMINF - rmin * inv;
    float zmax = QMAXF - rmax * inv;
    float emin = 128.0f + fabsf(rmin * inv);
    float emax = 127.0f + fabsf(rmax * inv);
    float zp0 = (emin < emax) ? zmin : zmax;
    int zp = (int)rintf(zp0);
    if (zp0 < QMINF) zp = -128;
    if (zp0 > QMAXF) zp = 127;

    if (lane == 0) {
        scales[row] = scale;
        zps[row]    = zp;
        sums[row]   = (int)rintf(vsum * inv);
    }

    const float fzp = (float)zp;
    int* qrow = (int*)(q + (size_t)row * K);
#pragma unroll
    for (int j = 0; j < 8; ++j) {
        if (j >= nch) break;
        float4 v = vals[j];
        int q0, q1, q2, q3;
        {
            float t;
            t = fminf(fmaxf(rintf(v.x * inv) + fzp, QMINF), QMAXF); q0 = (int)t;
            t = fminf(fmaxf(rintf(v.y * inv) + fzp, QMINF), QMAXF); q1 = (int)t;
            t = fminf(fmaxf(rintf(v.z * inv) + fzp, QMINF), QMAXF); q2 = (int)t;
            t = fminf(fmaxf(rintf(v.w * inv) + fzp, QMINF), QMAXF); q3 = (int)t;
        }
        int packed = (q0 & 0xff) | ((q1 & 0xff) << 8) | ((q2 & 0xff) << 16) | (q3 << 24);
        qrow[lane + (j << 5)] = packed;
    }
}

__global__ __launch_bounds__(256) void igemm_wmma_kernel(
    const int8_t* __restrict__ qx, const int8_t* __restrict__ qw,
    const float* __restrict__ scales, const int* __restrict__ zps,
    const int* __restrict__ sums,
    const float* __restrict__ bias, const float* __restrict__ w_scales,
    const int* __restrict__ w_zeros, const int* __restrict__ w_sums,
    float* __restrict__ out, int M, int N, int K)
{
    __shared__ char lds[4 * TILE_BYTES];

    auto ldsA = [&](int buf) -> char* { return lds + buf * TILE_BYTES; };
    auto ldsB = [&](int buf) -> char* { return lds + (2 + buf) * TILE_BYTES; };

    const int blockM = blockIdx.y * BM;
    const int blockN = blockIdx.x * BN;
    const int lane = threadIdx.x & 31;
    const int wave = threadIdx.x >> 5;
    const int wm = wave & 3;
    const int wn = wave >> 2;
    const int half = lane >> 4;
    const int l15 = lane & 15;

    auto issue_stage = [&](int buf, int k0) {
        const int8_t* ga_base = qx + (size_t)blockM * K + k0;
        const int8_t* gb_base = qw + (size_t)blockN * K + k0;
        char* la_base = ldsA(buf);
        char* lb_base = ldsB(buf);
        const int t = threadIdx.x;
#pragma unroll
        for (int i = 0; i < 2; ++i) {
            int idx = (i << 8) + t;
            int r = idx >> 2;
            int c = (idx & 3) << 4;
            uint64_t ga = (uint64_t)(uintptr_t)(ga_base + (size_t)r * K + c);
            uint32_t la = (uint32_t)(uintptr_t)(la_base + r * LDST + c);
            asm volatile("global_load_async_to_lds_b128 %0, %1, off"
                         :: "v"(la), "v"(ga) : "memory");
            uint64_t gb = (uint64_t)(uintptr_t)(gb_base + (size_t)r * K + c);
            uint32_t lb = (uint32_t)(uintptr_t)(lb_base + r * LDST + c);
            asm volatile("global_load_async_to_lds_b128 %0, %1, off"
                         :: "v"(lb), "v"(gb) : "memory");
        }
    };

    v8i zero = {};
    v8i acc[2][4];
#pragma unroll
    for (int mi = 0; mi < 2; ++mi)
#pragma unroll
        for (int ni = 0; ni < 4; ++ni) acc[mi][ni] = zero;

    const int KT = K / BK;
    issue_stage(0, 0);

    for (int kt = 0; kt < KT; ++kt) {
        const int cur = kt & 1;
        if (kt + 1 < KT) {
            issue_stage(cur ^ 1, (kt + 1) * BK);
            asm volatile("s_wait_asynccnt 0x4" ::: "memory");
        } else {
            asm volatile("s_wait_asynccnt 0x0" ::: "memory");
        }
        __syncthreads();

        v8i bfrag[4];
#pragma unroll
        for (int ni = 0; ni < 4; ++ni) {
            const char* bp = ldsB(cur) + (size_t)(wn * 64 + ni * 16 + l15) * LDST + half * 16;
            int4 lo = *(const int4*)bp;
            int4 hi = *(const int4*)(bp + 32);
            v8i b = zero;
            b[0] = lo.x; b[1] = lo.y; b[2] = lo.z; b[3] = lo.w;
            b[4] = hi.x; b[5] = hi.y; b[6] = hi.z; b[7] = hi.w;
            bfrag[ni] = b;
        }
#pragma unroll
        for (int mi = 0; mi < 2; ++mi) {
            const char* ap = ldsA(cur) + (size_t)(wm * 32 + mi * 16 + l15) * LDST + half * 16;
            int2 a0 = *(const int2*)(ap);
            int2 a1 = *(const int2*)(ap + 8);
            int2 a2 = *(const int2*)(ap + 32);
            int2 a3 = *(const int2*)(ap + 40);
            v8i a = zero;
            a[0] = a0.x; a[1] = a0.y; a[2] = a1.x; a[3] = a1.y;
            a[4] = a2.x; a[5] = a2.y; a[6] = a3.x; a[7] = a3.y;
#pragma unroll
            for (int ni = 0; ni < 4; ++ni) {
                acc[mi][ni] = __builtin_amdgcn_wmma_i32_16x16x64_iu8(
                    true, a, true, bfrag[ni], acc[mi][ni], false, false);
            }
        }
        __syncthreads();
    }

    const int mrow0 = blockM + wm * 32;
#pragma unroll
    for (int mi = 0; mi < 2; ++mi) {
        const int mbase = mrow0 + mi * 16 + half * 8;
        float rs[8]; int rzp[8]; int rsm[8];
#pragma unroll
        for (int v = 0; v < 8; ++v) {
            rs[v]  = scales[mbase + v];
            rzp[v] = zps[mbase + v];
            rsm[v] = sums[mbase + v];
        }
#pragma unroll
        for (int ni = 0; ni < 4; ++ni) {
            const int n = blockN + wn * 64 + ni * 16 + l15;
            const float wsc = w_scales[n];
            const int   wz  = w_zeros[n];
            const int   wsm = w_sums[n];
            const float bb  = bias[n];
#pragma unroll
            for (int v = 0; v < 8; ++v) {
                int dint = acc[mi][ni][v];
                int temp = rzp[v] * wsm + rsm[v] * wz;
                out[(size_t)(mbase + v) * N + n] =
                    bb + (float)(dint - temp) * (rs[v] * wsc);
            }
        }
    }
}

extern "C" void kernel_launch(void* const* d_in, const int* in_sizes, int n_in,
                              void* d_out, int out_size, void* d_ws, size_t ws_size,
                              hipStream_t stream)
{
    (void)n_in; (void)out_size; (void)ws_size;
    const float*  inp      = (const float*)d_in[0];
    const int8_t* qweight  = (const int8_t*)d_in[1];
    const float*  bias     = (const float*)d_in[2];
    const float*  w_scales = (const float*)d_in[3];
    const int*    w_zeros  = (const int*)d_in[4];
    const int*    w_sums   = (const int*)d_in[5];

    const int N = in_sizes[2];
    const int K = in_sizes[1] / N;
    const int M = in_sizes[0] / K;

    int8_t* qx     = (int8_t*)d_ws;
    float*  scales = (float*)((char*)d_ws + (size_t)M * K);
    int*    zps    = (int*)(scales + M);
    int*    sums   = (int*)(zps + M);

    quant_rows_kernel<<<dim3((M + 7) / 8), dim3(256), 0, stream>>>(
        inp, qx, scales, zps, sums, M, K);

    igemm_wmma_kernel<<<dim3(N / BN, M / BM), dim3(256), 0, stream>>>(
        qx, qweight, scales, zps, sums, bias, w_scales, w_zeros, w_sums,
        (float*)d_out, M, N, K);
}
